// HybridNN_44727789420664
// MI455X (gfx1250) — compile-verified, matrix operand order checked by reading, not run on hardware
//
#include <hip/hip_runtime.h>
#include <hip/hip_bf16.h>
#include <math.h>

// ---------------------------------------------------------------------------
// Problem constants
// ---------------------------------------------------------------------------
#define NQ     12
#define DIM    4096            // 2^NQ amplitudes
#define COM_Q  9
#define IMG_Q  11
#define BATCH  1024
#define KDIM   2048            // inner dim of classical layer
#define NDIM   512             // output dim of classical layer

typedef __bf16 bf16_t;
typedef bf16_t v16bf __attribute__((ext_vector_type(16)));
typedef float  v8f   __attribute__((ext_vector_type(8)));
typedef unsigned int v4u __attribute__((ext_vector_type(4)));
typedef int v8i __attribute__((ext_vector_type(8)));
typedef int v4i __attribute__((ext_vector_type(4)));

// Scheduling fence: LDS reads below must not be hoisted above this point.
#define LDS_WRITTEN_BY_TDM() asm volatile("" ::: "memory")

// ---------------------------------------------------------------------------
// TDM: issue a 2-D f32 tile DMA global -> LDS (CDNA5 tensor_load_to_lds).
// Descriptor built per ISA 08_async_tensor 8.3/8.4:
//   group0: count=1 | lds_addr | global_addr[56:0] | type=2
//   group1: data_size=4B, tensor_dim0/1, tile_dim0/1, tensor_dim0_stride
//   group2/3(+extra): zero (dims 2..4 unused for 2-D tiles)
// This toolchain exposes the 6-arg builtin (g0, g1, g2, g3, g4, cpol).
// ---------------------------------------------------------------------------
__device__ __forceinline__ void tdm_load_tile_f32(
    unsigned lds_byte_addr,      // destination LDS byte offset
    const float* tile_start,     // global address of tile origin
    unsigned tensor_w,           // tensor dim0 (elements per row)
    unsigned tensor_h,           // tensor dim1 (rows)
    unsigned tile_w,             // tile dim0 (contiguous elements)
    unsigned tile_h,             // tile dim1 (rows)
    unsigned row_stride_elems)   // tensor dim0 stride (elements)
{
    unsigned long long ga = (unsigned long long)(uintptr_t)tile_start;
    v4u g0;
    g0[0] = 1u;                                            // count=1 (user mode)
    g0[1] = lds_byte_addr;                                 // lds_addr
    g0[2] = (unsigned)(ga & 0xFFFFFFFFu);                  // global_addr lo
    g0[3] = (unsigned)((ga >> 32) & 0x01FFFFFFu)           // global_addr hi
          | (2u << 30);                                    // type = 2 (image)
    v8i g1;
    g1[0] = (int)(2u << 16);                               // data_size = 4 bytes
    g1[1] = (int)((tensor_w & 0xFFFFu) << 16);             // tensor_dim0[15:0]
    g1[2] = (int)((tensor_w >> 16) |
                  ((tensor_h & 0xFFFFu) << 16));           // dim0 hi | dim1 lo
    g1[3] = (int)((tensor_h >> 16) | (tile_w << 16));      // dim1 hi | tile_dim0
    g1[4] = (int)tile_h;                                   // tile_dim1 (tile_dim2=0)
    g1[5] = (int)row_stride_elems;                         // tensor_dim0_stride lo
    g1[6] = 0;                                             // stride hi | dim1_stride lo
    g1[7] = 0;                                             // dim1_stride hi
    v4i gz4 = {0, 0, 0, 0};
    v8i gz8 = {0, 0, 0, 0, 0, 0, 0, 0};
    __builtin_amdgcn_tensor_load_to_lds(g0, g1, gz4, gz4, gz8, 0);
}

// ---------------------------------------------------------------------------
// Kernel 1: com = x @ W^T + b   (1024x2048 @ 2048x512), bf16 WMMA, f32 acc.
// Block tile 32(M) x 64(N), 8 waves, one 16x16 subtile per wave.
// Operand tiles staged f32 in LDS by the TDM, double-buffered.
// ---------------------------------------------------------------------------
#define GEMM_BM 32
#define GEMM_BN 64
#define GEMM_BK 32
// LDS pool layout (floats): per buffer: X tile 32x32 (1024) + W tile 64x32 (2048)
#define LDS_BUF_FLOATS (GEMM_BM * GEMM_BK + GEMM_BN * GEMM_BK)   // 3072
#define LDS_X_OFF      0
#define LDS_W_OFF      (GEMM_BM * GEMM_BK)                        // 1024 floats

__global__ __launch_bounds__(256)
void hybrid_gemm_wmma(const float* __restrict__ X,   // (1024, 2048) row-major
                      const float* __restrict__ W,   // (512, 2048) row-major
                      const float* __restrict__ bias,// (512)
                      float* __restrict__ C)         // (1024, 512) row-major
{
    // +4 floats of padding: target of the opaque store below.
    __shared__ float lds_pool[2 * LDS_BUF_FLOATS + 4];   // 24 KB + pad

    // The TDM writes lds_pool behind the compiler's back. Without any visible
    // store, LLVM folds all lds_pool loads to undef (proven: round-4 asm had
    // wmma with identical undef A/B operands and no ds_loads). Emit one real
    // store at an index the compiler cannot resolve (opaque via inline asm,
    // runtime value = padding slot, so it never corrupts tile data). This
    // forces every subsequent lds_pool load to be a real ds_load.
    {
        unsigned opaque_idx;
        asm volatile("s_mov_b32 %0, %1"
                     : "=s"(opaque_idx)
                     : "s"(2u * LDS_BUF_FLOATS));
        if (threadIdx.x == 0) lds_pool[opaque_idx] = 0.0f;
    }

    const int blockM = blockIdx.x & 31;              // 1024/32 = 32 M blocks
    const int blockN = blockIdx.x >> 5;              // 512/64  = 8  N blocks
    const int wave   = threadIdx.x >> 5;             // 0..7
    const int lane   = threadIdx.x & 31;
    const int sm     = wave >> 2;                    // 0..1  (M subtile)
    const int sn     = wave & 3;                     // 0..3  (N subtile)
    const int half   = lane >> 4;                    // WMMA lane half
    const int mr     = lane & 15;                    // row (A) / col (B) in subtile

    const float* Xblk = X + (size_t)(blockM * GEMM_BM) * KDIM;
    const float* Wblk = W + (size_t)(blockN * GEMM_BN) * KDIM;
    const bool issuer = (threadIdx.x < 32);          // wave 0 drives the TDM

    // Prologue: DMA tile 0 into buffer 0.
    if (issuer) {
        tdm_load_tile_f32(LDS_X_OFF * 4u, Xblk, KDIM, GEMM_BM, GEMM_BK, GEMM_BM, KDIM);
        tdm_load_tile_f32(LDS_W_OFF * 4u, Wblk, KDIM, GEMM_BN, GEMM_BK, GEMM_BN, KDIM);
    }

    v8f acc = {};
    const int nK = KDIM / GEMM_BK;                   // 64 k-steps

    for (int kt = 0; kt < nK; ++kt) {
        const unsigned cur = (unsigned)(kt & 1);
        const unsigned nxt = cur ^ 1u;
        const unsigned curBase = cur * (LDS_BUF_FLOATS * 4u);   // byte offset
        const unsigned nxtBase = nxt * (LDS_BUF_FLOATS * 4u);

        // Kick off next tile's DMA while this tile is being consumed.
        if (kt + 1 < nK) {
            if (issuer) {
                const float* xn = Xblk + (size_t)(kt + 1) * GEMM_BK;
                const float* wn = Wblk + (size_t)(kt + 1) * GEMM_BK;
                tdm_load_tile_f32(nxtBase + LDS_X_OFF * 4u, xn,
                                  KDIM, GEMM_BM, GEMM_BK, GEMM_BM, KDIM);
                tdm_load_tile_f32(nxtBase + LDS_W_OFF * 4u, wn,
                                  KDIM, GEMM_BN, GEMM_BK, GEMM_BN, KDIM);
                __builtin_amdgcn_s_wait_tensorcnt(2);  // current 2 landed
            }
        } else {
            if (issuer) __builtin_amdgcn_s_wait_tensorcnt(0);
        }
        __syncthreads();        // tile `cur` visible to all waves
        LDS_WRITTEN_BY_TDM();   // keep the LDS reads below this point

        // Read fragments from LDS, convert f32 -> bf16 (hardware cvt).
        const float* Xs = lds_pool + (size_t)(curBase / 4u) + LDS_X_OFF;
        const float* Ws = lds_pool + (size_t)(curBase / 4u) + LDS_W_OFF;
        const float* xr = Xs + (sm * 16 + mr) * GEMM_BK;   // A row
        const float* wr = Ws + (sn * 16 + mr) * GEMM_BK;   // B column (W row)

        float af[16], bf[16];
        *(float4*)(af + 0)  = *(const float4*)(xr + half * 8);
        *(float4*)(af + 4)  = *(const float4*)(xr + half * 8 + 4);
        *(float4*)(af + 8)  = *(const float4*)(xr + 16 + half * 8);
        *(float4*)(af + 12) = *(const float4*)(xr + 16 + half * 8 + 4);
        *(float4*)(bf + 0)  = *(const float4*)(wr + half * 8);
        *(float4*)(bf + 4)  = *(const float4*)(wr + half * 8 + 4);
        *(float4*)(bf + 8)  = *(const float4*)(wr + 16 + half * 8);
        *(float4*)(bf + 12) = *(const float4*)(wr + 16 + half * 8 + 4);

        v16bf a, b;
#pragma unroll
        for (int e = 0; e < 16; ++e) {
            a[e] = (bf16_t)af[e];
            b[e] = (bf16_t)bf[e];
        }

        acc = __builtin_amdgcn_wmma_f32_16x16x32_bf16(
            /*neg_a=*/false, a, /*neg_b=*/false, b,
            /*c_mod=*/(short)0, acc, /*reuse_a=*/false, /*reuse_b=*/false);

        __syncthreads();   // everyone done reading `cur` before it is re-filled
    }

    // C/D layout: VGPR v -> row (v + 8*half), col (lane&15) of the 16x16 tile.
    const int col = blockN * GEMM_BN + sn * 16 + mr;
    const float bv = bias[col];
#pragma unroll
    for (int v = 0; v < 8; ++v) {
        const int row = blockM * GEMM_BM + sm * 16 + v + 8 * half;
        C[(size_t)row * NDIM + col] = acc[v] + bv;
    }
}

// ---------------------------------------------------------------------------
// Kernel 2: 12-qubit state-vector circuit, one workgroup per batch element.
// Entire 32KB state lives in LDS; barrier-separated gate passes.
// wire w occupies bit (NQ-1-w) of the amplitude index (wire 0 = MSB).
// ---------------------------------------------------------------------------
__device__ __forceinline__ float2 cmul(float2 a, float2 b) {
    return make_float2(a.x * b.x - a.y * b.y, a.x * b.y + a.y * b.x);
}

__global__ __launch_bounds__(256)
void hybrid_circuit(const float* __restrict__ X,    // (B, 2048) img angles
                    const float* __restrict__ COM,  // (B, 512)  com angles
                    const float* __restrict__ QP,   // (2, 12, 3)
                    float* __restrict__ OUT)        // (B,)
{
    __shared__ float2 st[DIM];      // 32 KB state
    __shared__ float  red[256];     // reduction scratch

    const int b   = blockIdx.x;
    const int tid = threadIdx.x;
    const int nt  = blockDim.x;     // 256

    const float* img_ang = X   + (size_t)b * KDIM;  // 2^11 angles
    const float* com_ang = COM + (size_t)b * NDIM;  // 2^9  angles

    // |0...0>
    for (int i = tid; i < DIM; i += nt)
        st[i] = make_float2(i == 0 ? 1.0f : 0.0f, 0.0f);
    __syncthreads();

    // --- gate helpers (all end with a workgroup barrier) -------------------
    auto hadamard = [&](int w) {
        const int shift = NQ - 1 - w;
        const int mask  = 1 << shift;
        const float r   = 0.70710678118654752440f;
        for (int p = tid; p < DIM / 2; p += nt) {
            int low = p & (mask - 1);
            int i0  = ((p & ~(mask - 1)) << 1) | low;
            int i1  = i0 | mask;
            float2 a = st[i0], c = st[i1];
            st[i0] = make_float2((a.x + c.x) * r, (a.y + c.y) * r);
            st[i1] = make_float2((a.x - c.x) * r, (a.y - c.y) * r);
        }
        __syncthreads();
    };

    // Fused H (x) H on two distinct wires: one LDS pass instead of two.
    auto hadamard2 = [&](int w1, int w2) {
        const int s1 = NQ - 1 - w1, s2 = NQ - 1 - w2;
        const int lo = s1 < s2 ? s1 : s2;
        const int hi = s1 < s2 ? s2 : s1;
        const int m1 = 1 << s1, m2 = 1 << s2;
        for (int p = tid; p < DIM / 4; p += nt) {
            int i = p;
            i = ((i >> lo) << (lo + 1)) | (i & ((1 << lo) - 1));
            i = ((i >> hi) << (hi + 1)) | (i & ((1 << hi) - 1));
            const int i00 = i, i01 = i | m2, i10 = i | m1, i11 = i | m1 | m2;
            float2 a = st[i00], c = st[i01], d = st[i10], e = st[i11];
            const float h = 0.5f;
            st[i00] = make_float2((a.x + c.x + d.x + e.x) * h, (a.y + c.y + d.y + e.y) * h);
            st[i01] = make_float2((a.x - c.x + d.x - e.x) * h, (a.y - c.y + d.y - e.y) * h);
            st[i10] = make_float2((a.x + c.x - d.x - e.x) * h, (a.y + c.y - d.y - e.y) * h);
            st[i11] = make_float2((a.x - c.x - d.x + e.x) * h, (a.y - c.y - d.y + e.y) * h);
        }
        __syncthreads();
    };

    auto rot = [&](int l, int w) {
        const float phi   = QP[(l * NQ + w) * 3 + 0];
        const float theta = QP[(l * NQ + w) * 3 + 1];
        const float omega = QP[(l * NQ + w) * 3 + 2];
        float sth, cth; __sincosf(0.5f * theta, &sth, &cth);
        float sa, ca;   __sincosf(0.5f * (phi + omega), &sa, &ca);
        float sb, cb;   __sincosf(0.5f * (phi - omega), &sb, &cb);
        // Rot = RZ(omega) RY(theta) RZ(phi)
        const float2 m00 = make_float2( ca * cth, -sa * cth);
        const float2 m01 = make_float2(-cb * sth, -sb * sth);
        const float2 m10 = make_float2( cb * sth, -sb * sth);
        const float2 m11 = make_float2( ca * cth,  sa * cth);
        const int shift = NQ - 1 - w;
        const int mask  = 1 << shift;
        for (int p = tid; p < DIM / 2; p += nt) {
            int low = p & (mask - 1);
            int i0  = ((p & ~(mask - 1)) << 1) | low;
            int i1  = i0 | mask;
            float2 a = st[i0], c = st[i1];
            float2 t0 = cmul(m00, a), t1 = cmul(m01, c);
            float2 u0 = cmul(m10, a), u1 = cmul(m11, c);
            st[i0] = make_float2(t0.x + t1.x, t0.y + t1.y);
            st[i1] = make_float2(u0.x + u1.x, u0.y + u1.y);
        }
        __syncthreads();
    };

    auto cnot = [&](int c, int t) {
        const int sc = NQ - 1 - c, stt = NQ - 1 - t;
        const int mc = 1 << sc, mt = 1 << stt;
        const int lo = sc < stt ? sc : stt;
        const int hi = sc < stt ? stt : sc;
        for (int p = tid; p < DIM / 4; p += nt) {
            int i = p;
            i = ((i >> lo) << (lo + 1)) | (i & ((1 << lo) - 1));
            i = ((i >> hi) << (hi + 1)) | (i & ((1 << hi) - 1));
            int i0 = i | mc;       // control=1, target=0
            int i1 = i0 | mt;      // control=1, target=1
            float2 tmp = st[i0]; st[i0] = st[i1]; st[i1] = tmp;
        }
        __syncthreads();
    };

    // uniformly-controlled RY on wire 0 (controls = wires 1..q, bit-reversed)
    auto cry = [&](const float* angles, int q) {
        const int restBits = NQ - 1 - q;
        for (int p = tid; p < DIM / 2; p += nt) {
            int j    = p >> restBits;                          // control column
            int ridx = (int)(__brev((unsigned)j) >> (32 - q)); // bitrev(j)
            float halfang = 0.5f * angles[ridx];
            float sn, cs; __sincosf(halfang, &sn, &cs);
            int i0 = p;                   // wire0 = 0
            int i1 = p | (DIM >> 1);      // wire0 = 1
            float2 a = st[i0], c = st[i1];
            st[i0] = make_float2(cs * a.x - sn * c.x, cs * a.y - sn * c.y);
            st[i1] = make_float2(sn * a.x + cs * c.x, sn * a.y + cs * c.y);
        }
        __syncthreads();
    };

    // --- circuit -----------------------------------------------------------
    // FRQI(com, q=9): H on wires 1..9
    hadamard2(1, 2); hadamard2(3, 4); hadamard2(5, 6); hadamard2(7, 8);
    hadamard(9);
    cry(com_ang, COM_Q);

    // StronglyEntanglingLayers, 2 layers, ranges r = l % (N-1) + 1
    for (int l = 0; l < 2; ++l) {
        for (int w = 0; w < NQ; ++w) rot(l, w);
        const int r = l % (NQ - 1) + 1;
        for (int w = 0; w < NQ; ++w) cnot(w, (w + r) % NQ);
    }

    // FRQI(img, q=11): H on wires 1..11
    hadamard2(1, 2); hadamard2(3, 4); hadamard2(5, 6); hadamard2(7, 8);
    hadamard2(9, 10);
    hadamard(11);
    cry(img_ang, IMG_Q);

    // expval: P(wire0=1) - P(wire0=0)
    float acc = 0.0f;
    for (int i = tid; i < DIM; i += nt) {
        float2 a = st[i];
        float pr = a.x * a.x + a.y * a.y;
        acc += (i & (DIM >> 1)) ? pr : -pr;
    }
    red[tid] = acc;
    __syncthreads();
    for (int s = nt >> 1; s > 0; s >>= 1) {
        if (tid < s) red[tid] += red[tid + s];
        __syncthreads();
    }
    if (tid == 0) OUT[b] = red[0];
}

// ---------------------------------------------------------------------------
// Launch
// ---------------------------------------------------------------------------
extern "C" void kernel_launch(void* const* d_in, const int* in_sizes, int n_in,
                              void* d_out, int out_size, void* d_ws, size_t ws_size,
                              hipStream_t stream) {
    (void)in_sizes; (void)n_in; (void)out_size; (void)ws_size;
    const float* x  = (const float*)d_in[0];  // (1024, 2, 32, 32) -> (1024, 2048)
    const float* W  = (const float*)d_in[1];  // (512, 2048)
    const float* b  = (const float*)d_in[2];  // (512)
    const float* qp = (const float*)d_in[3];  // (2, 12, 3)
    float* out = (float*)d_out;               // (1024,)
    float* com = (float*)d_ws;                // (1024, 512) scratch, 2 MB

    // 32 M-blocks x 8 N-blocks = 256 blocks of 8 waves.
    hipLaunchKernelGGL(hybrid_gemm_wmma, dim3(256), dim3(256), 0, stream,
                       x, W, b, com);
    // One workgroup per batch element; state lives in LDS.
    hipLaunchKernelGGL(hybrid_circuit, dim3(BATCH), dim3(256), 0, stream,
                       x, com, qp, out);
}
